// GCN_Module_6476810682380
// MI455X (gfx1250) — matrix operand order repaired and checked by reading, not compiled, not run on hardware
//
#include <hip/hip_runtime.h>
#include <hip/hip_bf16.h>

// ---------------------------------------------------------------------------
// GCN module for MI455X (gfx1250), wave32.
//   - all GEMMs via v_wmma_f32_16x16x32_bf16 (f32 accumulate)
//   - LDS tile staging via Tensor Data Mover (tensor_load_to_lds, TENSORcnt),
//     double-buffered: next K-step tiles DMA while current tiles feed WMMA
//   - TDM issue branches scalarized (readfirstlane) -> uniform s_cbranch
//   - 2x2 register blocking per wave in the dominant GEMM (4 WMMA per 4 frag
//     loads), K-step 64 to halve barrier/DMA frequency
//   - operands pre-transposed at bf16-conversion time so every LDS tile is a
//     plain row-major 2D tile (TDM cannot transpose)
// B=512, N=64, NFG=1024, NFR=256
// ---------------------------------------------------------------------------

typedef __attribute__((ext_vector_type(16))) __bf16    bf16x16;
typedef __attribute__((ext_vector_type(8)))  float     f32x8;
typedef __attribute__((ext_vector_type(4)))  unsigned  uint32x4;
typedef __attribute__((ext_vector_type(8)))  unsigned  uint32x8;

#define BB   512
#define NN   64
#define NFG  1024
#define NFR  256

__device__ __forceinline__ unsigned short f2bf(float f) {
    unsigned u = __builtin_bit_cast(unsigned, f);
    unsigned r = u + 0x7fffu + ((u >> 16) & 1u);   // round to nearest even
    return (unsigned short)(r >> 16);
}

__device__ __forceinline__ f32x8 wmma_bf16(bf16x16 a, bf16x16 b, f32x8 c) {
    return __builtin_amdgcn_wmma_f32_16x16x32_bf16(
        false, a, false, b, (short)0, c, false, false);
}

// A fragment: tile [M][K] row-major, K contiguous (ISA 7.12.2 16-bit A layout).
__device__ __forceinline__ bf16x16 fragA(const unsigned short* p, int ld) {
    int l = threadIdx.x & 31;
    int m = l & 15, h = l >> 4;
    const unsigned short* r = p + m * ld + h * 8;
    union { unsigned u[8]; bf16x16 v; } x;
#pragma unroll
    for (int j = 0; j < 4; ++j) x.u[j]     = *(const unsigned*)(r + 2 * j);
#pragma unroll
    for (int j = 0; j < 4; ++j) x.u[4 + j] = *(const unsigned*)(r + 16 + 2 * j);
    return x.v;
}

// B fragment: tile [N][K], K contiguous (ISA 7.12.2 16-bit B layout).
__device__ __forceinline__ bf16x16 fragB(const unsigned short* p, int ld) {
    int l = threadIdx.x & 31;
    int n = l & 15, h = l >> 4;
    const unsigned short* r = p + n * ld + h * 8;
    union { unsigned u[8]; bf16x16 v; } x;
#pragma unroll
    for (int j = 0; j < 4; ++j) x.u[j]     = *(const unsigned*)(r + 2 * j);
#pragma unroll
    for (int j = 0; j < 4; ++j) x.u[4 + j] = *(const unsigned*)(r + 16 + 2 * j);
    return x.v;
}

// ---------------------------------------------------------------------------
// Tensor Data Mover: DMA a 2D bf16 tile (tile_w x tile_h, row stride
// stride_elems) from global memory into LDS at lds_dst (packed). D# per CDNA5
// ISA ch.8. Issue from ONE wave (EXEC ignored; one DMA per instruction);
// completion tracked with TENSORcnt, in-order per wave -> double buffering
// via s_wait_tensorcnt(<ops in flight for next step>).
// ---------------------------------------------------------------------------
__device__ __forceinline__ void tdm_load_2d(const unsigned short* lds_dst,
                                            const unsigned short* gsrc,
                                            unsigned tile_w, unsigned tile_h,
                                            unsigned stride_elems) {
    unsigned lds_off = (unsigned)(unsigned long long)(uintptr_t)lds_dst; // addr[31:0]
    unsigned long long ga = (unsigned long long)(uintptr_t)gsrc;
    const unsigned td = 1u << 20;        // "infinite" tensor extent (no OOB clip)
    uint32x4 g0;
    g0[0] = 1u;                                          // count=1, user D#
    g0[1] = lds_off;                                     // lds_addr
    g0[2] = (unsigned)(ga & 0xffffffffull);              // global_addr[95:64]
    g0[3] = (unsigned)((ga >> 32) & 0x1ffffffull)        // global_addr[120:96]
          | (2u << 30);                                  // type=2 ("image")
    uint32x8 g1;
    g1[0] = 1u << 16;                                    // mask=0, data_size=1 (2B)
    g1[1] = (td & 0xffffu) << 16;                        // tensor_dim0[15:0]
    g1[2] = (td >> 16) | ((td & 0xffffu) << 16);         // td0 hi | tensor_dim1 lo
    g1[3] = (td >> 16) | (tile_w << 16);                 // td1 hi | tile_dim0
    g1[4] = tile_h;                                      // tile_dim1 (tile_dim2=0)
    g1[5] = stride_elems;                                // tensor_dim0_stride[31:0]
    g1[6] = 0;
    g1[7] = 0;
    asm volatile("tensor_load_to_lds %0, %1" :: "s"(g0), "s"(g1) : "memory");
}

// ---------------------------------------------------------------------------
// Kernel 0a: straight f32 -> bf16 conversion
// ---------------------------------------------------------------------------
__global__ void cvt_bf16_kernel(const float* __restrict__ x,
                                unsigned short* __restrict__ y, int n) {
    int i = blockIdx.x * blockDim.x + threadIdx.x;
    int stride = gridDim.x * blockDim.x;
    for (; i < n; i += stride) y[i] = f2bf(x[i]);
}

// Kernel 0b: transpose+convert, power-of-two dims: in [R][C] f32 -> [C][R] bf16
__global__ void cvtT_kernel(const float* __restrict__ x,
                            unsigned short* __restrict__ y,
                            int R, int cshift /* C = 1<<cshift */) {
    int i = blockIdx.x * blockDim.x + threadIdx.x;
    int total = R << cshift;
    int stride = gridDim.x * blockDim.x;
    int cmask = (1 << cshift) - 1;
    for (; i < total; i += stride) {
        int r = i >> cshift, c = i & cmask;
        y[(size_t)c * R + r] = f2bf(x[i]);
    }
}

// Kernel 0c: batched transpose feats [B][N][F] f32 -> [B][F][N] bf16
__global__ void cvtT_feats_kernel(const float* __restrict__ x,
                                  unsigned short* __restrict__ y) {
    int i = blockIdx.x * blockDim.x + threadIdx.x;
    int stride = gridDim.x * blockDim.x;
    const int total = BB * NN * NFG;
    for (; i < total; i += stride) {
        int b = i >> 16;                   // / (NN*NFG)
        int rem = i & 0xffff;
        int n = rem >> 10, f = rem & 1023;
        y[((size_t)b * NFG + f) * NN + n] = f2bf(x[i]);
    }
}

// ---------------------------------------------------------------------------
// Kernel 1: theta/phi fused GEMM  [32768x1024]x[1024x256] (x2, shared A tile)
// 64x64 block tile, 8 waves, 2 (16x16) subtiles each per output matrix.
// K-step 64 (2 WMMA-deep per staging round); TDM double-buffered, 3 tiles.
// ---------------------------------------------------------------------------
__global__ __launch_bounds__(256) void gemm_theta_phi(
    const unsigned short* __restrict__ Xb,    // feats bf16 [32768][1024]
    const unsigned short* __restrict__ WtT,   // W_theta^T bf16 [256][1024]
    const unsigned short* __restrict__ WpT,   // W_phi^T   bf16 [256][1024]
    const float* __restrict__ bt, const float* __restrict__ bp,
    unsigned short* __restrict__ Th,          // theta bf16 [32768][256]
    unsigned short* __restrict__ Ph)          // phi   bf16 [32768][256]
{
    __shared__ __attribute__((aligned(16))) unsigned short lA[2][64 * 64];  // 16 KB
    __shared__ __attribute__((aligned(16))) unsigned short lBt[2][64 * 64]; // 16 KB
    __shared__ __attribute__((aligned(16))) unsigned short lBp[2][64 * 64]; // 16 KB

    const int t    = threadIdx.x;
    const int wave = __builtin_amdgcn_readfirstlane(t >> 5);  // uniform
    const int lane = t & 31;
    const int mt = blockIdx.y;   // 0..511
    const int nt = blockIdx.x;   // 0..3
    const int sm = wave >> 1;    // M-subtile (0..3)

    const unsigned short* gA  = Xb  + (size_t)(mt * 64) * NFG;
    const unsigned short* gBt = WtT + (size_t)(nt * 64) * NFG;
    const unsigned short* gBp = WpT + (size_t)(nt * 64) * NFG;

    f32x8 acct[2] = {};
    f32x8 accp[2] = {};

    if (wave == 0) {                       // prime buffer 0
        tdm_load_2d(lA[0],  gA,  64, 64, NFG);
        tdm_load_2d(lBt[0], gBt, 64, 64, NFG);
        tdm_load_2d(lBp[0], gBp, 64, 64, NFG);
    }

    for (int kk = 0; kk < NFG; kk += 64) {
        const int cur = (kk >> 6) & 1;
        if (wave == 0) {
            if (kk + 64 < NFG) {           // prefetch next K-step into other buf
                tdm_load_2d(lA[cur ^ 1],  gA  + kk + 64, 64, 64, NFG);
                tdm_load_2d(lBt[cur ^ 1], gBt + kk + 64, 64, 64, NFG);
                tdm_load_2d(lBp[cur ^ 1], gBp + kk + 64, 64, 64, NFG);
                __builtin_amdgcn_s_wait_tensorcnt(3);   // current 3 done
            } else {
                __builtin_amdgcn_s_wait_tensorcnt(0);
            }
        }
        __syncthreads();                   // buf[cur] ready for all waves

#pragma unroll
        for (int ks = 0; ks < 64; ks += 32) {
            bf16x16 a = fragA(&lA[cur][sm * 16 * 64 + ks], 64);
#pragma unroll
            for (int i = 0; i < 2; ++i) {
                int sn = (wave * 2 + i) & 3;
                bf16x16 b_t = fragB(&lBt[cur][sn * 16 * 64 + ks], 64);
                bf16x16 b_p = fragB(&lBp[cur][sn * 16 * 64 + ks], 64);
                acct[i] = wmma_bf16(a, b_t, acct[i]);
                accp[i] = wmma_bf16(a, b_p, accp[i]);
            }
        }
        __syncthreads();                   // done reading before buf reuse
    }

    // Epilogue: +bias, emit bf16 for downstream WMMA consumers.
#pragma unroll
    for (int i = 0; i < 2; ++i) {
        int sn  = (wave * 2 + i) & 3;
        int col = nt * 64 + sn * 16 + (lane & 15);
        float bth = bt[col], bph = bp[col];
#pragma unroll
        for (int r = 0; r < 8; ++r) {
            int row = mt * 64 + sm * 16 + (lane >> 4) * 8 + r;
            Th[(size_t)row * NFR + col] = f2bf(acct[i][r] + bth);
            Ph[(size_t)row * NFR + col] = f2bf(accp[i][r] + bph);
        }
    }
}

// ---------------------------------------------------------------------------
// Kernel 2: per-batch fused attention:
//   sim = theta*phi^T/16 -> distance mask -> f32 softmax (writes Rel)
//   agg = rel * feats (bf16 WMMA), feats chunks double-buffered via TDM
// ---------------------------------------------------------------------------
__global__ __launch_bounds__(256) void attention_kernel(
    const float* __restrict__ boxes,          // [512][64][4]
    const unsigned short* __restrict__ Th,    // [32768][256]
    const unsigned short* __restrict__ Ph,    // [32768][256]
    const unsigned short* __restrict__ XbT,   // feats^T bf16 [512][1024][64]
    float* __restrict__ Rel,                  // [512][64][64]  (output 2)
    unsigned short* __restrict__ Agg)         // [32768][1024] bf16
{
    __shared__ float px[NN], py[NN];
    __shared__ __attribute__((aligned(16))) float sim[NN * NN];             // 16 KB
    __shared__ __attribute__((aligned(16))) unsigned short relb[NN * NN];   //  8 KB
    __shared__ __attribute__((aligned(16))) unsigned short ft[2][128 * NN]; // 32 KB

    const int b    = blockIdx.x;
    const int t    = threadIdx.x;
    const int wave = __builtin_amdgcn_readfirstlane(t >> 5);
    const int lane = t & 31;
    const int sm = wave >> 1;

    const unsigned short* xb = XbT + (size_t)b * NFG * NN;   // [f][m'] rows

    if (wave == 0)                          // prime feats chunk 0 early: overlap
        tdm_load_2d(ft[0], xb, 64, 128, 64);

    if (t < NN) {
        const float* bx = boxes + (size_t)(b * NN + t) * 4;
        px[t] = (bx[0] + bx[2]) * 0.5f;
        py[t] = (bx[1] + bx[3]) * 0.5f;
    }
    __syncthreads();

    // ---- sim = Th_b * Ph_b^T * (1/sqrt(256)); fragments direct from global
    {
        const unsigned short* tb = Th + (size_t)b * NN * NFR;  // [i][r]
        const unsigned short* pb = Ph + (size_t)b * NN * NFR;  // [m][r] == [n][k]
        f32x8 acc[2] = {};
        for (int kk = 0; kk < NFR; kk += 32) {
            bf16x16 a = fragA(tb + (size_t)(sm * 16) * NFR + kk, NFR);
#pragma unroll
            for (int i = 0; i < 2; ++i) {
                int sn = (wave * 2 + i) & 3;
                bf16x16 bb = fragB(pb + (size_t)(sn * 16) * NFR + kk, NFR);
                acc[i] = wmma_bf16(a, bb, acc[i]);
            }
        }
        // scale + distance mask (dist^2 > (0.2*157)^2 = 985.96)
#pragma unroll
        for (int i = 0; i < 2; ++i) {
            int sn = (wave * 2 + i) & 3;
            int j  = sn * 16 + (lane & 15);
            float pxj = px[j], pyj = py[j];
#pragma unroll
            for (int r = 0; r < 8; ++r) {
                int ii = sm * 16 + (lane >> 4) * 8 + r;
                float dx = px[ii] - pxj, dy = py[ii] - pyj;
                float v = acc[i][r] * 0.0625f;
                if (dx * dx + dy * dy > 985.96f) v = -__builtin_inff();
                sim[ii * NN + j] = v;
            }
        }
    }
    __syncthreads();

    // ---- row softmax (diag never masked -> finite max) ----
    if (t < NN) {
        float mx = -__builtin_inff();
        for (int j = 0; j < NN; ++j) mx = fmaxf(mx, sim[t * NN + j]);
        float s = 0.f;
        for (int j = 0; j < NN; ++j) s += __expf(sim[t * NN + j] - mx);
        float inv = 1.f / s;
        for (int j = 0; j < NN; ++j) {
            float r = __expf(sim[t * NN + j] - mx) * inv;
            Rel[((size_t)b * NN + t) * NN + j] = r;   // exact f32 output
            relb[t * NN + j] = f2bf(r);               // bf16 A for agg GEMM
        }
    }
    __syncthreads();

    // ---- agg = rel(64x64) * feats(64x1024), 128-wide chunks, double-buffered
    for (int nc = 0; nc < 8; ++nc) {
        const int cur = nc & 1;
        if (wave == 0) {
            if (nc + 1 < 8) {
                tdm_load_2d(ft[cur ^ 1], xb + (size_t)(nc + 1) * 128 * NN,
                            64, 128, 64);
                __builtin_amdgcn_s_wait_tensorcnt(1);
            } else {
                __builtin_amdgcn_s_wait_tensorcnt(0);
            }
        }
        __syncthreads();

#pragma unroll
        for (int ms = 0; ms < 4; ++ms) {
            f32x8 acc = {};
#pragma unroll
            for (int kk = 0; kk < NN; kk += 32) {
                bf16x16 a  = fragA(&relb[ms * 16 * NN + kk], NN);
                bf16x16 bb = fragB(&ft[cur][(wave * 16) * NN + kk], NN);
                acc = wmma_bf16(a, bb, acc);
            }
            int col = nc * 128 + wave * 16 + (lane & 15);
#pragma unroll
            for (int r = 0; r < 8; ++r) {
                int row = b * NN + ms * 16 + (lane >> 4) * 8 + r;
                Agg[(size_t)row * NFG + col] = f2bf(acc[r]);
            }
        }
        __syncthreads();
    }
}

// ---------------------------------------------------------------------------
// Kernel 3: out = relu(agg * W_gcn)   [32768x1024]x[1024x1024] -> f32
// Dominant GEMM (68.7 GFLOP). Block tile 128x64, 8 waves in 4(M)x2(N) grid,
// 2x2 subtiles per wave (4 WMMA per 4 fragment loads), K-step 64,
// double-buffered TDM (2 ops in flight).
// ---------------------------------------------------------------------------
__global__ __launch_bounds__(256) void gemm_out_kernel(
    const unsigned short* __restrict__ Ag,    // [32768][1024] bf16
    const unsigned short* __restrict__ WgT,   // W_gcn^T bf16 [1024][1024]
    float* __restrict__ Out)                  // [32768][1024] f32
{
    __shared__ __attribute__((aligned(16))) unsigned short lA[2][128 * 64]; // 32 KB
    __shared__ __attribute__((aligned(16))) unsigned short lB[2][64 * 64];  // 16 KB

    const int t    = threadIdx.x;
    const int wave = __builtin_amdgcn_readfirstlane(t >> 5);
    const int lane = t & 31;
    const int mt = blockIdx.y;   // 0..255  (128-row tiles)
    const int nt = blockIdx.x;   // 0..15   (64-col tiles)
    const int wm = wave >> 1;    // 0..3  -> rows wm*32..+32
    const int wn = wave & 1;     // 0..1  -> cols wn*32..+32

    const unsigned short* gA = Ag  + (size_t)(mt * 128) * NFG;
    const unsigned short* gB = WgT + (size_t)(nt * 64) * NFG;

    f32x8 acc[2][2] = {};

    if (wave == 0) {
        tdm_load_2d(lA[0], gA, 64, 128, NFG);
        tdm_load_2d(lB[0], gB, 64, 64,  NFG);
    }

    for (int kk = 0; kk < NFG; kk += 64) {
        const int cur = (kk >> 6) & 1;
        if (wave == 0) {
            if (kk + 64 < NFG) {
                tdm_load_2d(lA[cur ^ 1], gA + kk + 64, 64, 128, NFG);
                tdm_load_2d(lB[cur ^ 1], gB + kk + 64, 64, 64,  NFG);
                __builtin_amdgcn_s_wait_tensorcnt(2);
            } else {
                __builtin_amdgcn_s_wait_tensorcnt(0);
            }
        }
        __syncthreads();

#pragma unroll
        for (int ks = 0; ks < 64; ks += 32) {
            bf16x16 a0 = fragA(&lA[cur][(wm * 32 +  0) * 64 + ks], 64);
            bf16x16 a1 = fragA(&lA[cur][(wm * 32 + 16) * 64 + ks], 64);
            bf16x16 b0 = fragB(&lB[cur][(wn * 32 +  0) * 64 + ks], 64);
            bf16x16 b1 = fragB(&lB[cur][(wn * 32 + 16) * 64 + ks], 64);
            acc[0][0] = wmma_bf16(a0, b0, acc[0][0]);
            acc[0][1] = wmma_bf16(a0, b1, acc[0][1]);
            acc[1][0] = wmma_bf16(a1, b0, acc[1][0]);
            acc[1][1] = wmma_bf16(a1, b1, acc[1][1]);
        }
        __syncthreads();
    }

#pragma unroll
    for (int im = 0; im < 2; ++im) {
#pragma unroll
        for (int in = 0; in < 2; ++in) {
            int col = nt * 64 + wn * 32 + in * 16 + (lane & 15);
#pragma unroll
            for (int r = 0; r < 8; ++r) {
                int row = mt * 128 + wm * 32 + im * 16 + (lane >> 4) * 8 + r;
                Out[(size_t)row * NFG + col] = fmaxf(acc[im][in][r], 0.f);
            }
        }
    }
}

// ---------------------------------------------------------------------------
// Host launcher
// ---------------------------------------------------------------------------
extern "C" void kernel_launch(void* const* d_in, const int* in_sizes, int n_in,
                              void* d_out, int out_size, void* d_ws, size_t ws_size,
                              hipStream_t stream) {
    const float* feats = (const float*)d_in[0];   // [512,64,1024]
    const float* boxes = (const float*)d_in[1];   // [512,64,4]
    const float* Wth   = (const float*)d_in[2];   // [1024,256]
    const float* bth   = (const float*)d_in[3];   // [256]
    const float* Wph   = (const float*)d_in[4];   // [1024,256]
    const float* bph   = (const float*)d_in[5];   // [256]
    const float* Wg    = (const float*)d_in[6];   // [1024,1024]

    const size_t nFeat = (size_t)BB * NN * NFG;   // 33,554,432
    const size_t nWtp  = (size_t)NFG * NFR;       //    262,144
    const size_t nWg   = (size_t)NFG * NFG;       //  1,048,576
    const size_t nTh   = (size_t)BB * NN * NFR;   //  8,388,608

    unsigned short* Xb  = (unsigned short*)d_ws;  // feats bf16 [B*N][F]
    unsigned short* XbT = Xb  + nFeat;            // feats^T bf16 [B][F][N]
    unsigned short* WtT = XbT + nFeat;            // W_theta^T [256][1024]
    unsigned short* WpT = WtT + nWtp;             // W_phi^T   [256][1024]
    unsigned short* WgT = WpT + nWtp;             // W_gcn^T   [1024][1024]
    unsigned short* Th  = WgT + nWg;              // theta bf16 [32768][256]
    unsigned short* Ph  = Th  + nTh;              // phi   bf16 [32768][256]
    unsigned short* Agg = Ph  + nTh;              // agg   bf16 [32768][1024]

    float* Out = (float*)d_out;                   // [512,64,1024]
    float* Rel = Out + nFeat;                     // [512,64,64]

    // 0) bf16 conversions (+ transposes so every GEMM tile is row-major [*, K])
    cvt_bf16_kernel  <<<16384, 256, 0, stream>>>(feats, Xb, (int)nFeat);
    cvtT_feats_kernel<<<16384, 256, 0, stream>>>(feats, XbT);
    cvtT_kernel      <<<1024,  256, 0, stream>>>(Wth, WtT, NFG, 8);   // C=256
    cvtT_kernel      <<<1024,  256, 0, stream>>>(Wph, WpT, NFG, 8);   // C=256
    cvtT_kernel      <<<4096,  256, 0, stream>>>(Wg,  WgT, NFG, 10);  // C=1024

    // 1) theta/phi fused GEMM
    gemm_theta_phi<<<dim3(NFR / 64, (BB * NN) / 64), 256, 0, stream>>>(
        Xb, WtT, WpT, bth, bph, Th, Ph);

    // 2) per-batch attention + aggregation
    attention_kernel<<<BB, 256, 0, stream>>>(boxes, Th, Ph, XbT, Rel, Agg);

    // 3) final GEMM + ReLU
    gemm_out_kernel<<<dim3(NFG / 64, (BB * NN) / 128), 256, 0, stream>>>(
        Agg, WgT, Out);
}
